// TransformerLM_46608985096888
// MI455X (gfx1250) — matrix operand order repaired and checked by reading, not compiled, not run on hardware
//
#include <hip/hip_runtime.h>
#include <hip/hip_bf16.h>
#include <math.h>

typedef __attribute__((ext_vector_type(16))) _Float16 v16h;
typedef __attribute__((ext_vector_type(8)))  _Float16 v8h;
typedef __attribute__((ext_vector_type(8)))  float    v8f;

#define V_   32000
#define D_   1024
#define NH_  16
#define HD_  64
#define L_   2
#define E_   8
#define H_   2048
#define B_   2
#define T_   1024
#define NTOK (B_*T_)

union V16U { v16h v; v8h h8[2]; };

__device__ inline v16h make_v16(const _Float16* lo, const _Float16* hi) {
    V16U u;
    u.h8[0] = *reinterpret_cast<const v8h*>(lo);
    u.h8[1] = *reinterpret_cast<const v8h*>(hi);
    return u.v;
}

// ---------------------------------------------------------------------------
// Tiled WMMA GEMM: C[M,N] = A[M,K] * B, fp32 in/out, f16 WMMA compute.
//   BT==0 : B is [K,N] row-major.   BT==1 : B is [N,K] row-major (B^T).
//   MODE 0: C = acc
//   MODE 1: C = acc + resid  (residual add; C may alias resid)
//   MODE 2: C += rowscale[row*rss] * acc   (MoE combine-accumulate)
// Block tile 64x128x32, 256 threads (8 waves), each wave = 2x2 WMMA tiles.
// Register-pipelined staging: tile k+1 global->VGPR while tile k computes
// from LDS; tile k+2 prefetched (global_prefetch_b8).
// ---------------------------------------------------------------------------
#define BM 64
#define BN 128
#define BK 32
#define KP 8   // LDS row pad (halves); row pitch 40h = 80B (16B multiple)

template<int BT, int MODE>
__global__ __launch_bounds__(256) void gemm_wmma(
    const float* __restrict__ A, const float* __restrict__ Bm,
    float* __restrict__ C, const float* __restrict__ resid,
    const float* __restrict__ rowscale, int rss,
    int M, int N, int Kd)
{
    __shared__ __align__(16) _Float16 Ash[BM][BK + KP];
    __shared__ __align__(16) _Float16 Bsh[BN][BK + KP];

    const int tid  = threadIdx.x;
    const int m0   = blockIdx.y * BM;
    const int n0   = blockIdx.x * BN;
    const int lane = tid & 31, wave = tid >> 5;
    const int hh   = lane >> 4, ln = lane & 15;
    const int wm   = wave >> 2, wn = wave & 3;

    v8f acc[2][2];
#pragma unroll
    for (int a = 0; a < 2; a++)
#pragma unroll
        for (int b = 0; b < 2; b++)
#pragma unroll
            for (int i = 0; i < 8; i++) acc[a][b][i] = 0.0f;

    // staging assignments
    const int arow = tid >> 2;           // 0..63
    const int aks  = (tid & 3) * 8;      // 0,8,16,24
    const int bn0  = (BT == 0) ? (tid & 127) : (tid >> 1);
    const int bk0  = (BT == 0) ? ((tid >> 7) * 16) : ((tid & 1) * 16);

    const float* aptr = A + (size_t)(m0 + arow) * Kd + aks;
    const float* bptr = (BT == 0) ? (Bm + (size_t)bk0 * N + n0 + bn0)
                                  : (Bm + (size_t)(n0 + bn0) * Kd + bk0);
    const size_t bstep = (BT == 0) ? ((size_t)BK * N) : (size_t)BK;

    float ar[8], br[16];
    auto fetch = [&](int kt) {
        const float* ap = aptr + (size_t)kt * BK;
#pragma unroll
        for (int j = 0; j < 8; j++) ar[j] = ap[j];
        const float* bp = bptr + (size_t)kt * bstep;
        if (BT == 0) {
#pragma unroll
            for (int kk = 0; kk < 16; kk++) br[kk] = bp[(size_t)kk * N];
        } else {
#pragma unroll
            for (int kk = 0; kk < 16; kk++) br[kk] = bp[kk];
        }
    };

    const int nk = Kd / BK;
    fetch(0);

    for (int kt = 0; kt < nk; kt++) {
        // stage current tile (fp32 -> f16) into LDS
#pragma unroll
        for (int j = 0; j < 8; j++) Ash[arow][aks + j] = (_Float16)ar[j];
#pragma unroll
        for (int kk = 0; kk < 16; kk++) Bsh[bn0][bk0 + kk] = (_Float16)br[kk];
        __syncthreads();

        // prefetch tile kt+2, fetch tile kt+1 into registers (overlaps WMMA)
        if (kt + 2 < nk) {
            __builtin_prefetch(aptr + (size_t)(kt + 2) * BK, 0, 3);
            __builtin_prefetch(bptr + (size_t)(kt + 2) * bstep, 0, 3);
        }
        if (kt + 1 < nk) fetch(kt + 1);

        v16h af[2], bf[2];
#pragma unroll
        for (int ms = 0; ms < 2; ms++) {
            const _Float16* base = &Ash[wm * 32 + ms * 16 + ln][0];
            af[ms] = make_v16(base + hh * 8, base + 16 + hh * 8);
        }
#pragma unroll
        for (int ns = 0; ns < 2; ns++) {
            const _Float16* base = &Bsh[wn * 32 + ns * 16 + ln][0];
            bf[ns] = make_v16(base + hh * 8, base + 16 + hh * 8);
        }
#pragma unroll
        for (int ms = 0; ms < 2; ms++)
#pragma unroll
            for (int ns = 0; ns < 2; ns++)
                acc[ms][ns] = __builtin_amdgcn_wmma_f32_16x16x32_f16(
                    false, af[ms], false, bf[ns], (short)0, acc[ms][ns], false, false);
        __syncthreads();
    }

#pragma unroll
    for (int ms = 0; ms < 2; ms++)
#pragma unroll
        for (int ns = 0; ns < 2; ns++)
#pragma unroll
            for (int i = 0; i < 8; i++) {
                const int row = m0 + wm * 32 + ms * 16 + i + 8 * hh;
                const int col = n0 + wn * 32 + ns * 16 + ln;
                const size_t idx = (size_t)row * N + col;
                const float v = acc[ms][ns][i];
                if (MODE == 1)      C[idx] = v + resid[idx];
                else if (MODE == 2) C[idx] += rowscale[(size_t)row * rss] * v;
                else                C[idx] = v;
            }
}

// ---------------------------------------------------------------------------
// Flash attention: 1 wave = 16 queries x full head (HD=64). Causal, online
// softmax. q16/k16: [B*NH, T, HD] f16.  vT16: [B*NH, HD, T] f16.
// Scores and P*V both on v_wmma_f32_16x16x32_f16; P round-trips per-wave LDS
// to convert C layout -> A-fragment layout (s_wait_dscnt guards the RAW).
// ---------------------------------------------------------------------------
__global__ __launch_bounds__(128) void flash_attn(
    const _Float16* __restrict__ q16, const _Float16* __restrict__ k16,
    const _Float16* __restrict__ vT16, float* __restrict__ out)
{
    __shared__ __align__(16) _Float16 P[4][16][32];
    const int tid  = threadIdx.x;
    const int lane = tid & 31, wv = tid >> 5;
    const int hh   = lane >> 4, ln = lane & 15;
    const int bh   = blockIdx.x;               // b*NH + head
    const int b    = bh >> 4, head = bh & 15;
    const int qtile = blockIdx.y * 4 + wv;
    const int qbase = qtile * 16;

    const _Float16* qrow = q16 + ((size_t)bh * T_ + qbase + ln) * HD_;
    v16h aq[2];
    aq[0] = make_v16(qrow + hh * 8,      qrow + 16 + hh * 8);   // hd 0..31
    aq[1] = make_v16(qrow + 32 + hh * 8, qrow + 48 + hh * 8);   // hd 32..63

    float mi[8], li[8];
    v8f accO[4];
#pragma unroll
    for (int i = 0; i < 8; i++) { mi[i] = -1e30f; li[i] = 0.0f; }
#pragma unroll
    for (int t = 0; t < 4; t++)
#pragma unroll
        for (int i = 0; i < 8; i++) accO[t][i] = 0.0f;

    const int nkb = ((qbase + 15) >> 5) + 1;   // 32-key blocks, causal bound
    for (int kb = 0; kb < nkb; kb++) {
        v8f S[2];
#pragma unroll
        for (int s = 0; s < 2; s++) {
            const _Float16* krow = k16 + ((size_t)bh * T_ + kb * 32 + s * 16 + ln) * HD_;
            v16h b0 = make_v16(krow + hh * 8,      krow + 16 + hh * 8);       // hd 0..31
            v16h b1 = make_v16(krow + 32 + hh * 8, krow + 48 + hh * 8);  // hd 32..63
            v8f z;
#pragma unroll
            for (int i = 0; i < 8; i++) z[i] = 0.0f;
            z = __builtin_amdgcn_wmma_f32_16x16x32_f16(false, aq[0], false, b0, (short)0, z, false, false);
            z = __builtin_amdgcn_wmma_f32_16x16x32_f16(false, aq[1], false, b1, (short)0, z, false, false);
            S[s] = z;
        }
#pragma unroll
        for (int i = 0; i < 8; i++) {
            const int row = qbase + i + 8 * hh;
            const int c0 = kb * 32 + ln, c1 = c0 + 16;
            float s0 = S[0][i] * 0.125f;
            float s1 = S[1][i] * 0.125f;
            if (c0 > row) s0 = -1e30f;
            if (c1 > row) s1 = -1e30f;
            float vmax = fmaxf(s0, s1);
#pragma unroll
            for (int m = 1; m < 16; m <<= 1) vmax = fmaxf(vmax, __shfl_xor(vmax, m, 32));
            const float mnew = fmaxf(mi[i], vmax);
            const float p0 = __expf(s0 - mnew), p1 = __expf(s1 - mnew);
            float rs = p0 + p1;
#pragma unroll
            for (int m = 1; m < 16; m <<= 1) rs += __shfl_xor(rs, m, 32);
            const float corr = __expf(mi[i] - mnew);
            li[i] = li[i] * corr + rs;
            mi[i] = mnew;
#pragma unroll
            for (int t = 0; t < 4; t++) accO[t][i] *= corr;
            const int r = i + 8 * hh;
            P[wv][r][ln]      = (_Float16)p0;
            P[wv][r][16 + ln] = (_Float16)p1;
        }
        asm volatile("s_wait_dscnt 0" ::: "memory");
        const _Float16* prow = &P[wv][ln][0];
        v16h ap = make_v16(prow + hh * 8, prow + 16 + hh * 8);
#pragma unroll
        for (int t = 0; t < 4; t++) {
            const _Float16* vrow = vT16 + ((size_t)bh * HD_ + t * 16 + ln) * T_ + kb * 32;
            v16h bv = make_v16(vrow + hh * 8, vrow + 16 + hh * 8);
            accO[t] = __builtin_amdgcn_wmma_f32_16x16x32_f16(false, ap, false, bv, (short)0, accO[t], false, false);
        }
    }
#pragma unroll
    for (int t = 0; t < 4; t++)
#pragma unroll
        for (int i = 0; i < 8; i++) {
            const int row = qbase + i + 8 * hh;
            out[((size_t)b * T_ + row) * D_ + head * HD_ + t * 16 + ln] = accO[t][i] / li[i];
        }
}

// ---------------------------------------------------------------------------
// Small kernels
// ---------------------------------------------------------------------------
__global__ void embed_k(const int* __restrict__ tok, const float* __restrict__ emb,
                        float* __restrict__ h)
{
    const int n = blockIdx.x;
    const int tk = tok[n];
    const float4* s = (const float4*)(emb + (size_t)tk * D_);
    float4* d = (float4*)(h + (size_t)n * D_);
    d[threadIdx.x] = s[threadIdx.x];   // 256 threads * 4 = 1024
}

__global__ __launch_bounds__(256) void rmsnorm_k(const float* __restrict__ x,
                                                 const float* __restrict__ w,
                                                 float* __restrict__ o)
{
    __shared__ float red[256];
    const int row = blockIdx.x, tid = threadIdx.x;
    const float* xr = x + (size_t)row * D_;
    float s = 0.0f;
    for (int j = tid; j < D_; j += 256) { const float v = xr[j]; s += v * v; }
    red[tid] = s; __syncthreads();
    for (int st = 128; st > 0; st >>= 1) { if (tid < st) red[tid] += red[tid + st]; __syncthreads(); }
    const float rms = rsqrtf(red[0] / (float)D_ + 1e-6f);
    float* orow = o + (size_t)row * D_;
    for (int j = tid; j < D_; j += 256) orow[j] = xr[j] * rms * w[j];
}

// RoPE + fp32->f16 + [B,T,NH,HD] -> [B*NH,T,HD] repack (for q and k)
__global__ void rope_pack_k(const float* __restrict__ src, _Float16* __restrict__ dst)
{
    const int idx = blockIdx.x * blockDim.x + threadIdx.x;  // B*T*NH*(HD/2)
    if (idx >= B_ * T_ * NH_ * (HD_ / 2)) return;
    const int i    = idx & 31;
    const int hd2  = idx >> 5;
    const int head = hd2 & 15;
    const int bt   = hd2 >> 4;
    const int t    = bt & (T_ - 1);
    const int b    = bt >> 10;
    const float inv = __expf(-(float)(2 * i) * (1.0f / (float)HD_) * 9.210340371976184f); // ln(1e4)
    const float f = (float)t * inv;
    float sn, cs; __sincosf(f, &sn, &cs);
    const float* p = src + (size_t)bt * D_ + head * HD_ + 2 * i;
    const float xr = p[0], xi = p[1];
    const size_t ob = (((size_t)(b * NH_ + head)) * T_ + t) * HD_ + 2 * i;
    dst[ob]     = (_Float16)(xr * cs - xi * sn);
    dst[ob + 1] = (_Float16)(xr * sn + xi * cs);
}

// V: [B,T,D] fp32 -> [B*NH, HD, T] f16 (transposed for P*V B-fragments)
__global__ void v_pack_k(const float* __restrict__ v, _Float16* __restrict__ vT)
{
    const int idx = blockIdx.x * blockDim.x + threadIdx.x;  // B*T*D
    if (idx >= B_ * T_ * D_) return;
    const int d  = idx & (D_ - 1);
    const int bt = idx >> 10;
    const int t  = bt & (T_ - 1);
    const int b  = bt >> 10;
    const int head = d >> 6, hd = d & 63;
    vT[(((size_t)(b * NH_ + head)) * HD_ + hd) * T_ + t] = (_Float16)v[idx];
}

__global__ void silu_mul_k(const float* __restrict__ a, const float* __restrict__ b,
                           float* __restrict__ o, int n)
{
    const int i = blockIdx.x * blockDim.x + threadIdx.x;
    if (i < n) { const float x = a[i]; o[i] = (x / (1.0f + __expf(-x))) * b[i]; }
}

// Top-2 routing: 1 wave per token. combine[tok][e], plus aux-loss stats.
__global__ __launch_bounds__(256) void gate_k(const float* __restrict__ xf,
                                              const float* __restrict__ gw,
                                              float* __restrict__ combine,
                                              float* __restrict__ psum,
                                              float* __restrict__ cnt)
{
    const int lane = threadIdx.x & 31, wv = threadIdx.x >> 5;
    const int tokn = blockIdx.x * 8 + wv;
    float a[E_];
#pragma unroll
    for (int e = 0; e < E_; e++) a[e] = 0.0f;
    for (int d = lane; d < D_; d += 32) {
        const float xv = xf[(size_t)tokn * D_ + d];
#pragma unroll
        for (int e = 0; e < E_; e++) a[e] += xv * gw[d * E_ + e];
    }
#pragma unroll
    for (int e = 0; e < E_; e++)
#pragma unroll
        for (int m = 1; m < 32; m <<= 1) a[e] += __shfl_xor(a[e], m, 32);
    if (lane == 0) {
        float mx = a[0];
#pragma unroll
        for (int e = 1; e < E_; e++) mx = fmaxf(mx, a[e]);
        float p[E_], sum = 0.0f;
#pragma unroll
        for (int e = 0; e < E_; e++) { p[e] = __expf(a[e] - mx); sum += p[e]; }
#pragma unroll
        for (int e = 0; e < E_; e++) p[e] /= sum;
        int i1 = 0;
#pragma unroll
        for (int e = 1; e < E_; e++) if (p[e] > p[i1]) i1 = e;
        int i2 = (i1 == 0) ? 1 : 0;
#pragma unroll
        for (int e = 0; e < E_; e++) if (e != i1 && p[e] > p[i2]) i2 = e;
        const float wsum = p[i1] + p[i2] + 1e-8f;
#pragma unroll
        for (int e = 0; e < E_; e++)
            combine[(size_t)tokn * E_ + e] =
                (e == i1) ? p[i1] / wsum : ((e == i2) ? p[i2] / wsum : 0.0f);
#pragma unroll
        for (int e = 0; e < E_; e++) atomicAdd(&psum[e], p[e]);
        atomicAdd(&cnt[i1], 1.0f);
        atomicAdd(&cnt[i2], 1.0f);
    }
}

// stats layout: [0]=aux, [1..8]=psum, [9..16]=cnt
__global__ void aux_k(float* __restrict__ stats)
{
    float s = 0.0f;
    for (int e = 0; e < E_; e++)
        s += (stats[9 + e] / (float)NTOK) * (stats[1 + e] / (float)NTOK);
    stats[0] += (float)E_ * s;
    for (int e = 0; e < E_; e++) { stats[1 + e] = 0.0f; stats[9 + e] = 0.0f; }
}

__global__ void zero_k(float* __restrict__ p, int n)
{
    const int i = blockIdx.x * blockDim.x + threadIdx.x;
    if (i < n) p[i] = 0.0f;
}

__global__ void write_aux_k(const float* __restrict__ stats, float* __restrict__ out)
{
    out[0] = stats[0];
}

// ---------------------------------------------------------------------------
extern "C" void kernel_launch(void* const* d_in, const int* in_sizes, int n_in,
                              void* d_out, int out_size, void* d_ws, size_t ws_size,
                              hipStream_t stream)
{
    (void)in_sizes; (void)n_in; (void)out_size; (void)ws_size;
    const int*   tokens  = (const int*)d_in[0];
    const float* emb     = (const float*)d_in[1];
    const float* attn_nw = (const float*)d_in[2];
    const float* wq      = (const float*)d_in[3];
    const float* wk      = (const float*)d_in[4];
    const float* wvw     = (const float*)d_in[5];
    const float* wo      = (const float*)d_in[6];
    const float* ffn_nw  = (const float*)d_in[7];
    const float* gate_w  = (const float*)d_in[8];
    const float* w1      = (const float*)d_in[9];
    const float* w2      = (const float*)d_in[10];
    const float* w3      = (const float*)d_in[11];
    const float* fin_nw  = (const float*)d_in[12];
    float* logits = (float*)d_out;

    char* ws = (char*)d_ws;
    size_t off = 0;
    auto alloc = [&](size_t bytes) -> void* {
        void* p = ws + off;
        off += (bytes + 255) & ~(size_t)255;
        return p;
    };
    float*    h     = (float*)alloc((size_t)NTOK * D_ * 4);
    float*    hn    = (float*)alloc((size_t)NTOK * D_ * 4);
    float*    qb    = (float*)alloc((size_t)NTOK * D_ * 4);
    float*    kb    = (float*)alloc((size_t)NTOK * D_ * 4);
    float*    vb    = (float*)alloc((size_t)NTOK * D_ * 4);
    float*    aout  = (float*)alloc((size_t)NTOK * D_ * 4);
    float*    xf    = (float*)alloc((size_t)NTOK * D_ * 4);
    float*    h1    = (float*)alloc((size_t)NTOK * H_ * 4);
    float*    h3    = (float*)alloc((size_t)NTOK * H_ * 4);
    float*    h13   = (float*)alloc((size_t)NTOK * H_ * 4);
    _Float16* q16   = (_Float16*)alloc((size_t)B_ * NH_ * T_ * HD_ * 2);
    _Float16* k16   = (_Float16*)alloc((size_t)B_ * NH_ * T_ * HD_ * 2);
    _Float16* vT16  = (_Float16*)alloc((size_t)B_ * NH_ * T_ * HD_ * 2);
    float*    comb  = (float*)alloc((size_t)NTOK * E_ * 4);
    float*    stats = (float*)alloc(17 * 4);

    auto gemm = [&](const float* A, const float* Bm, float* C, int M, int N, int K,
                    int bt, int mode, const float* resid, const float* rsc, int rss) {
        dim3 grid(N / BN, M / BM);
        if (bt == 0 && mode == 0)
            gemm_wmma<0, 0><<<grid, 256, 0, stream>>>(A, Bm, C, resid, rsc, rss, M, N, K);
        else if (bt == 0 && mode == 1)
            gemm_wmma<0, 1><<<grid, 256, 0, stream>>>(A, Bm, C, resid, rsc, rss, M, N, K);
        else if (bt == 0 && mode == 2)
            gemm_wmma<0, 2><<<grid, 256, 0, stream>>>(A, Bm, C, resid, rsc, rss, M, N, K);
        else
            gemm_wmma<1, 0><<<grid, 256, 0, stream>>>(A, Bm, C, resid, rsc, rss, M, N, K);
    };

    zero_k<<<1, 32, 0, stream>>>(stats, 17);
    embed_k<<<NTOK, 256, 0, stream>>>(tokens, emb, h);

    for (int l = 0; l < L_; l++) {
        const float* wql = wq + (size_t)l * D_ * D_;
        const float* wkl = wk + (size_t)l * D_ * D_;
        const float* wvl = wvw + (size_t)l * D_ * D_;
        const float* wol = wo + (size_t)l * D_ * D_;

        // attention
        rmsnorm_k<<<NTOK, 256, 0, stream>>>(h, attn_nw + (size_t)l * D_, hn);
        gemm(hn, wql, qb, NTOK, D_, D_, 0, 0, nullptr, nullptr, 0);
        gemm(hn, wkl, kb, NTOK, D_, D_, 0, 0, nullptr, nullptr, 0);
        gemm(hn, wvl, vb, NTOK, D_, D_, 0, 0, nullptr, nullptr, 0);
        {
            const int nrope = B_ * T_ * NH_ * (HD_ / 2);
            rope_pack_k<<<(nrope + 255) / 256, 256, 0, stream>>>(qb, q16);
            rope_pack_k<<<(nrope + 255) / 256, 256, 0, stream>>>(kb, k16);
            const int nv = B_ * T_ * D_;
            v_pack_k<<<(nv + 255) / 256, 256, 0, stream>>>(vb, vT16);
        }
        flash_attn<<<dim3(B_ * NH_, T_ / 64), 128, 0, stream>>>(q16, k16, vT16, aout);
        gemm(aout, wol, h, NTOK, D_, D_, 0, 1, h, nullptr, 0);   // h = x + o

        // MoE
        rmsnorm_k<<<NTOK, 256, 0, stream>>>(h, ffn_nw + (size_t)l * D_, xf);
        gate_k<<<NTOK / 8, 256, 0, stream>>>(xf, gate_w + (size_t)l * D_ * E_,
                                             comb, stats + 1, stats + 9);
        aux_k<<<1, 1, 0, stream>>>(stats);
        for (int e = 0; e < E_; e++) {
            const float* w1e = w1 + ((size_t)l * E_ + e) * D_ * H_;
            const float* w3e = w3 + ((size_t)l * E_ + e) * D_ * H_;
            const float* w2e = w2 + ((size_t)l * E_ + e) * H_ * D_;
            gemm(xf, w1e, h1, NTOK, H_, D_, 0, 0, nullptr, nullptr, 0);
            gemm(xf, w3e, h3, NTOK, H_, D_, 0, 0, nullptr, nullptr, 0);
            const int nh = NTOK * H_;
            silu_mul_k<<<(nh + 255) / 256, 256, 0, stream>>>(h1, h3, h13, nh);
            // h += combine[:,e] * (h13 @ w2[e])
            gemm(h13, w2e, h, NTOK, D_, H_, 0, 2, nullptr, comb + e, E_);
        }
    }

    // final norm + logits into d_out (emb used transposed: bt=1)
    rmsnorm_k<<<NTOK, 256, 0, stream>>>(h, fin_nw, hn);
    gemm(hn, emb, logits, NTOK, V_, D_, 1, 0, nullptr, nullptr, 0);
    write_aux_k<<<1, 1, 0, stream>>>(stats, logits + (size_t)NTOK * V_);
}
